// MolOpt_47175920779845
// MI455X (gfx1250) — matrix operand order repaired and checked by reading, not compiled, not run on hardware
//
#include <hip/hip_runtime.h>
#include <hip/hip_bf16.h>

// ---------------------------------------------------------------------------
// Fused Sinkhorn-OT + tangent-embedding + MLP for MI455X (gfx1250, wave32).
// One workgroup (256 thr = 8 wave32) per molecule b:
//   1) G = X·Ref^T via v_wmma_f32_16x16x32_bf16  -> C[l,n] in LDS (f32)
//   2) 50 log-domain Sinkhorn iterations entirely in LDS (v_exp_f32 bound)
//   3) V = P^T·X via WMMA -> tg written to out + staged bf16 in LDS
//   4) MLP: h = leakyrelu(tg·w0^T+b0); delta = h·w1^T+b1  (both WMMA)
// ---------------------------------------------------------------------------

#define BMOL 1024
#define LMAX 64
#define NREF 64
#define DDIM 50
#define DPAD 64
#define HDIM 256
#define SEPS 0.1f
#define SITERS 50
#define NEGW (-1e30f)

typedef __attribute__((ext_vector_type(16))) __bf16 v16bf;
typedef __attribute__((ext_vector_type(8)))  __bf16 v8bf;
typedef __attribute__((ext_vector_type(8)))  float  v8f;

// A-frag (16x32 bf16, MxK): lane m=lane%16; lanes>=16 hold K +16.
// Two contiguous 8-half (16B) chunks at kb and kb+8.
__device__ __forceinline__ v16bf load_a_frag(const __bf16* base, int ld, int row, int kb) {
    const v8bf* p0 = (const v8bf*)(base + row * ld + kb);
    const v8bf* p1 = (const v8bf*)(base + row * ld + kb + 8);
    v8bf lo = *p0, hi = *p1;
    v16bf a;
#pragma unroll
    for (int j = 0; j < 8; ++j) { a[j] = lo[j]; a[j + 8] = hi[j]; }
    return a;
}

// B-frag (32x16 bf16, KxN) from a K-contiguous (transposed) LDS store:
// lane holds column n=lane%16, 16 contiguous K at kb (lanes>=16: +16).
__device__ __forceinline__ v16bf load_b_frag_lds(const __bf16* base, int ld, int col, int kb) {
    const v8bf* p0 = (const v8bf*)(base + col * ld + kb);
    v8bf lo = p0[0], hi = p0[1];
    v16bf b;
#pragma unroll
    for (int j = 0; j < 8; ++j) { b[j] = lo[j]; b[j + 8] = hi[j]; }
    return b;
}

// B-frag from a 50-float global row (ref / w0), zero-padded to K=64.
// kb in {0,16,32,48}; only kb==48 straddles the row end (elements 48,49).
// Rows are 200 B -> 8-B aligned, so float2 (b64) loads are safe.
__device__ __forceinline__ v16bf load_b_frag_row50(const float* __restrict__ rowptr, int kb) {
    v16bf b = {};
    if (kb < 48) {
#pragma unroll
        for (int c = 0; c < 8; ++c) {
            float2 f2 = *(const float2*)(rowptr + kb + 2 * c);
            b[2 * c]     = (__bf16)f2.x;
            b[2 * c + 1] = (__bf16)f2.y;
        }
    } else {
        float2 f2 = *(const float2*)(rowptr + 48);
        b[0] = (__bf16)f2.x;
        b[1] = (__bf16)f2.y;
    }
    return b;
}

// B-frag from a 256-float global row (w1): rows 1024 B -> float4 (b128) loads,
// always in-bounds (caller clamps the row); lanes with !valid get zeros
// (branch-free cndmask, keeps EXEC convergent around the WMMA).
__device__ __forceinline__ v16bf load_b_frag_row256(const float* __restrict__ rowptr, int kb, bool valid) {
    v16bf b;
#pragma unroll
    for (int c = 0; c < 4; ++c) {
        float4 f4 = *(const float4*)(rowptr + kb + 4 * c);
        b[4 * c]     = (__bf16)f4.x;
        b[4 * c + 1] = (__bf16)f4.y;
        b[4 * c + 2] = (__bf16)f4.z;
        b[4 * c + 3] = (__bf16)f4.w;
    }
    if (!valid) { v16bf z = {}; b = z; }
    return b;
}

__device__ __forceinline__ v8f wmma_bf16(v16bf a, v16bf b, v8f c) {
    return __builtin_amdgcn_wmma_f32_16x16x32_bf16(false, a, false, b, (short)0, c, false, false);
}

__global__ __launch_bounds__(256)
void molopt_fused_kernel(const float* __restrict__ emb,   // [B,L,D]
                         const int*   __restrict__ lengths,
                         const float* __restrict__ ref,   // [N,D]
                         const float* __restrict__ w0,    // [H,D]
                         const float* __restrict__ b0,    // [H]
                         const float* __restrict__ w1,    // [D,H]
                         const float* __restrict__ b1,    // [D]
                         float*       __restrict__ out)   // tg [B*N,D] ++ delta [B*N,D]
{
    // ---- LDS carve (43.5 KB peak, phase-overlaid) ----
    __shared__ __attribute__((aligned(16))) unsigned char smem[43520];
    __bf16* sX  = (__bf16*)(smem);              // [64][64]   phase1 (A of G)
    __bf16* sXt = (__bf16*)(smem + 8192);       // [64][64]   phase1 (B of V, K-contig)
    float*  sC  = (float* )(smem + 16384);      // [64][65]   cost matrix
    __bf16* sPt = (__bf16*)(smem + 33024);      // [64][72]   P^T bf16
    float*  fv   = (float*)(smem + 42240);      // [64]
    float*  gv   = (float*)(smem + 42496);      // [64]
    float*  x2s  = (float*)(smem + 42752);      // [64]
    float*  r2s  = (float*)(smem + 43008);      // [64]
    float*  loga = (float*)(smem + 43264);      // [64]
    // phase-2 overlays (safe: separated by barriers)
    __bf16* sTg = (__bf16*)(smem);              // [64][64]   tangent embedding bf16
    __bf16* sH  = (__bf16*)(smem + 8192);       // [64][256]  hidden activations bf16

    const int b    = blockIdx.x;
    const int t    = threadIdx.x;
    const int lane = t & 31;
    const int wv   = t >> 5;
    const int hiA  = (lane & 16) ? 16 : 0;   // A-frag K adjust for lanes 16..31
    const int hiB  = (lane & 16) ? 16 : 0;   // B-frag K adjust for lanes 16..31
    const int cN   = lane & 15;              // C/D column, A row, B column
    const int len  = lengths[b];

    // Warm L2 with the shared MLP weights (global_prefetch_b8).
    if (wv == 0) {
        __builtin_prefetch(w0 + (size_t)lane * ((HDIM * DDIM) / 32), 0, 3);
        __builtin_prefetch(w1 + (size_t)lane * ((DDIM * HDIM) / 32), 0, 3);
    }

    // ---- Phase 0: scalars + stage X (masked, bf16, row-major + transposed) ----
    if (t < 64) {
        fv[t] = 0.0f; gv[t] = 0.0f;
        loga[t] = (t < len) ? -__logf((float)len) : NEGW;
        float s = 0.0f;
        const float* rp = ref + t * DDIM;
#pragma unroll
        for (int c = 0; c < 25; ++c) {
            float2 f2 = *(const float2*)(rp + 2 * c);
            s += f2.x * f2.x + f2.y * f2.y;
        }
        r2s[t] = s;
    }
    {
        // one thread per (row l, 16-col chunk d0): vector loads, no atomics
        const int l  = t >> 2;
        const int d0 = (t & 3) * 16;
        float vals[16];
#pragma unroll
        for (int j = 0; j < 16; ++j) vals[j] = 0.0f;
        if (l < len) {
            const float* rp = emb + ((size_t)b * LMAX + l) * DDIM;
            if (d0 < 48) {
#pragma unroll
                for (int c = 0; c < 8; ++c) {
                    float2 f2 = *(const float2*)(rp + d0 + 2 * c);
                    vals[2 * c]     = f2.x;
                    vals[2 * c + 1] = f2.y;
                }
            } else {
                float2 f2 = *(const float2*)(rp + 48);
                vals[0] = f2.x;
                vals[1] = f2.y;
            }
        }
        float s = 0.0f;
#pragma unroll
        for (int j = 0; j < 16; ++j) {
            sX [l * DPAD + d0 + j]   = (__bf16)vals[j];
            sXt[(d0 + j) * DPAD + l] = (__bf16)vals[j];
            s += vals[j] * vals[j];
        }
        s += __shfl_xor(s, 1, 32);
        s += __shfl_xor(s, 2, 32);
        if ((t & 3) == 0) x2s[l] = s;   // sum over all 64 padded cols == sum over D
    }
    __syncthreads();

    // ---- Phase 1: C[l,n] = x2 + r2 - 2*(X · Ref^T)  (16 tiles, 2/wave) ----
#pragma unroll
    for (int rep = 0; rep < 2; ++rep) {
        int tile = wv + rep * 8;
        int tm = tile >> 2, tn = tile & 3;
        v8f acc = {};
#pragma unroll
        for (int ks = 0; ks < 2; ++ks) {
            v16bf a  = load_a_frag(sX, DPAD, tm * 16 + cN, ks * 32 + hiA);
            v16bf bb = load_b_frag_row50(ref + (size_t)(tn * 16 + cN) * DDIM, ks * 32 + hiB);
            acc = wmma_bf16(a, bb, acc);
        }
        int n = tn * 16 + cN;
#pragma unroll
        for (int r = 0; r < 8; ++r) {
            int m = tm * 16 + r + ((lane & 16) ? 8 : 0);
            sC[m * 65 + n] = x2s[m] + r2s[n] - 2.0f * acc[r];
        }
    }
    __syncthreads();

    // ---- Phase 2: 50 log-domain Sinkhorn iterations (4 lanes per row/col) ----
    const float invE = 1.0f / SEPS;
    const float LOGB = -4.15888308f;        // -log(64)
    const int rid = t >> 2;                 // row (f) / col (g) id, 0..63
    const int prt = t & 3;                  // 16-element slice
    for (int it = 0; it < SITERS; ++it) {
        // f[l] = eps*(log_a[l] - LSE_n (g[n]-C[l,n])/eps)
        {
            float vb[16], m = -3.4e38f;
#pragma unroll
            for (int j = 0; j < 16; ++j) {
                int n = prt * 16 + j;
                float v = (gv[n] - sC[rid * 65 + n]) * invE;
                vb[j] = v; m = fmaxf(m, v);
            }
            m = fmaxf(m, __shfl_xor(m, 1, 32));
            m = fmaxf(m, __shfl_xor(m, 2, 32));
            float s = 0.0f;
#pragma unroll
            for (int j = 0; j < 16; ++j) s += __expf(vb[j] - m);
            s += __shfl_xor(s, 1, 32);
            s += __shfl_xor(s, 2, 32);
            if (prt == 0) fv[rid] = SEPS * (loga[rid] - (m + __logf(s)));
        }
        __syncthreads();
        // g[n] = eps*(log_b - LSE_l (f[l]-C[l,n])/eps)
        {
            float vb[16], m = -3.4e38f;
#pragma unroll
            for (int j = 0; j < 16; ++j) {
                int l = prt * 16 + j;
                float v = (fv[l] - sC[l * 65 + rid]) * invE;
                vb[j] = v; m = fmaxf(m, v);
            }
            m = fmaxf(m, __shfl_xor(m, 1, 32));
            m = fmaxf(m, __shfl_xor(m, 2, 32));
            float s = 0.0f;
#pragma unroll
            for (int j = 0; j < 16; ++j) s += __expf(vb[j] - m);
            s += __shfl_xor(s, 1, 32);
            s += __shfl_xor(s, 2, 32);
            if (prt == 0) gv[rid] = SEPS * (LOGB - (m + __logf(s)));
        }
        __syncthreads();
    }

    // ---- Phase 3: P^T (bf16, K-contiguous over l) ----
#pragma unroll
    for (int j = 0; j < 16; ++j) {
        int n = prt * 16 + j;
        float pv = __expf((fv[rid] + gv[n] - sC[rid * 65 + n]) * invE);
        sPt[n * 72 + rid] = (__bf16)pv;
    }
    __syncthreads();

    // ---- Phase 4: V = P^T · X ; tg = (N*V - ref)/sqrt(N) -> out + sTg ----
#pragma unroll
    for (int rep = 0; rep < 2; ++rep) {
        int tile = wv + rep * 8;
        int tn = tile >> 2, td = tile & 3;
        v8f acc = {};
#pragma unroll
        for (int ks = 0; ks < 2; ++ks) {
            v16bf a  = load_a_frag(sPt, 72, tn * 16 + cN, ks * 32 + hiA);
            v16bf bb = load_b_frag_lds(sXt, DPAD, td * 16 + cN, ks * 32 + hiB);
            acc = wmma_bf16(a, bb, acc);
        }
        int d = td * 16 + cN;
        if (d < DDIM) {
#pragma unroll
            for (int r = 0; r < 8; ++r) {
                int n = tn * 16 + r + ((lane & 16) ? 8 : 0);
                float tgv = (64.0f * acc[r] - ref[(size_t)n * DDIM + d]) * 0.125f;
                out[((size_t)b * NREF + n) * DDIM + d] = tgv;
                sTg[n * DPAD + d] = (__bf16)tgv;
            }
        } else {
#pragma unroll
            for (int r = 0; r < 8; ++r) {
                int n = tn * 16 + r + ((lane & 16) ? 8 : 0);
                sTg[n * DPAD + d] = (__bf16)0.0f;
            }
        }
    }
    __syncthreads();

    // ---- Phase 5: h = leakyrelu(tg · w0^T + b0)   (64 tiles, 8/wave) ----
#pragma unroll
    for (int rep = 0; rep < 8; ++rep) {
        int tile = wv * 8 + rep;
        int tm = tile >> 4, th = tile & 15;
        v8f acc = {};
#pragma unroll
        for (int ks = 0; ks < 2; ++ks) {
            v16bf a  = load_a_frag(sTg, DPAD, tm * 16 + cN, ks * 32 + hiA);
            v16bf bb = load_b_frag_row50(w0 + (size_t)(th * 16 + cN) * DDIM, ks * 32 + hiB);
            acc = wmma_bf16(a, bb, acc);
        }
        int h = th * 16 + cN;
        float bias = b0[h];
#pragma unroll
        for (int r = 0; r < 8; ++r) {
            int m = tm * 16 + r + ((lane & 16) ? 8 : 0);
            float hvv = acc[r] + bias;
            hvv = (hvv > 0.0f) ? hvv : 0.01f * hvv;
            sH[m * HDIM + h] = (__bf16)hvv;
        }
    }
    __syncthreads();

    // ---- Phase 6: delta = h · w1^T + b1  (K=256, 8 k-steps) ----
    float* out2 = out + (size_t)BMOL * NREF * DDIM;
#pragma unroll
    for (int rep = 0; rep < 2; ++rep) {
        int tile = wv + rep * 8;
        int tm = tile >> 2, td = tile & 3;
        int d = td * 16 + cN;
        bool dok = (d < DDIM);
        const float* rp = w1 + (size_t)(dok ? d : (DDIM - 1)) * HDIM;  // clamped, in-bounds
        v8f acc = {};
#pragma unroll
        for (int ks = 0; ks < 8; ++ks) {
            v16bf a  = load_a_frag(sH, HDIM, tm * 16 + cN, ks * 32 + hiA);
            v16bf bb = load_b_frag_row256(rp, ks * 32 + hiB, dok);
            acc = wmma_bf16(a, bb, acc);
        }
        if (dok) {
            float bias = b1[d];
#pragma unroll
            for (int r = 0; r < 8; ++r) {
                int n = tm * 16 + r + ((lane & 16) ? 8 : 0);
                out2[((size_t)b * NREF + n) * DDIM + d] = acc[r] + bias;
            }
        }
    }
}

extern "C" void kernel_launch(void* const* d_in, const int* in_sizes, int n_in,
                              void* d_out, int out_size, void* d_ws, size_t ws_size,
                              hipStream_t stream) {
    const float* emb     = (const float*)d_in[0];
    const int*   lengths = (const int*)  d_in[1];
    const float* ref     = (const float*)d_in[2];
    const float* w0      = (const float*)d_in[3];
    const float* b0      = (const float*)d_in[4];
    const float* w1      = (const float*)d_in[5];
    const float* b1      = (const float*)d_in[6];
    float* out = (float*)d_out;
    (void)in_sizes; (void)n_in; (void)out_size; (void)d_ws; (void)ws_size;
    molopt_fused_kernel<<<dim3(BMOL), dim3(256), 0, stream>>>(emb, lengths, ref, w0, b0, w1, b1, out);
}
